// nh_spa_mapper_41824391528685
// MI455X (gfx1250) — matrix operand order repaired and checked by reading, not compiled, not run on hardware
//
#include <hip/hip_runtime.h>
#include <hip/hip_bf16.h>
#include <math.h>

#define BB    8
#define SS    8192
#define TT    8192
#define MD    16
#define FF    1024
#define OUTD  256
#define PEH   32
#define NH    16
#define HEADS 4
#define HD    (MD / HEADS)
#define SLOPE 0.2f
#define NROWS (BB * TT)

typedef __attribute__((ext_vector_type(16))) _Float16 v16h;
typedef __attribute__((ext_vector_type(8)))  _Float16 v8h;
typedef __attribute__((ext_vector_type(8)))  float    v8f;

__device__ __forceinline__ float lrelu(float a) {
    return a >= 0.0f ? a : SLOPE * a;
}

// ---------------------------------------------------------------------------
// Kernel 1: brute-force top-NH nearest neighbors (1 thread / target).
// ---------------------------------------------------------------------------
__global__ __launch_bounds__(256) void knn_kernel(
    const float* __restrict__ ct, const float* __restrict__ cs,
    int* __restrict__ idx)
{
    int t = blockIdx.x * blockDim.x + threadIdx.x;
    if (t >= TT) return;
    float tx = ct[t];
    float ty = ct[TT + t];
    float bd[NH];
    int   bi[NH];
#pragma unroll
    for (int i = 0; i < NH; i++) { bd[i] = 3.4e38f; bi[i] = 0; }
    for (int s = 0; s < SS; s++) {
        float dx = tx - cs[s];
        float dy = ty - cs[SS + s];
        float d  = dx * dx + dy * dy;
        if (d < bd[NH - 1]) {
            int p = NH - 1;
            while (p > 0 && bd[p - 1] > d) {
                bd[p] = bd[p - 1];
                bi[p] = bi[p - 1];
                p--;
            }
            bd[p] = d;
            bi[p] = s;
        }
    }
#pragma unroll
    for (int i = 0; i < NH; i++) idx[t * NH + i] = bi[i];
}

// ---------------------------------------------------------------------------
// Kernel 2: positional-encoding MLP (2 -> 32 -> 16), batch independent.
// ---------------------------------------------------------------------------
__global__ __launch_bounds__(256) void pe_kernel(
    const float* __restrict__ ct, const float* __restrict__ cs,
    const int* __restrict__ idx,
    const float* __restrict__ w1, const float* __restrict__ b1,
    const float* __restrict__ w2, const float* __restrict__ b2,
    float* __restrict__ pe)
{
    int g = blockIdx.x * blockDim.x + threadIdx.x;
    if (g >= TT * NH) return;
    int t = g >> 4;
    int j = g & 15;
    int s = idx[t * NH + j];
    float dx = cs[s]      - ct[t];
    float dy = cs[SS + s] - ct[TT + t];
    float h[PEH];
#pragma unroll
    for (int k = 0; k < PEH; k++)
        h[k] = lrelu(dx * w1[k] + dy * w1[PEH + k] + b1[k]);
#pragma unroll
    for (int o = 0; o < MD; o++) {
        float a = b2[o];
#pragma unroll
        for (int k = 0; k < PEH; k++) a += h[k] * w2[k * MD + o];
        pe[g * MD + o] = a;
    }
}

// ---------------------------------------------------------------------------
// Kernel 3: pack out_w1 / out_w2 into f16 WMMA B-fragment order:
//   K = kt*32 + (lane>>4)*8 + (i&7) + (i>>3)*16 , N = nt*16 + (lane&15)
//   stored at ((nt*Ktiles + kt)*32 + lane)*16 + i  -> one v16h load per lane.
// ---------------------------------------------------------------------------
__global__ __launch_bounds__(256) void pack_w_kernel(
    const float* __restrict__ w1, const float* __restrict__ w2,
    _Float16* __restrict__ w1p, _Float16* __restrict__ w2p)
{
    int g = blockIdx.x * blockDim.x + threadIdx.x;
    if (g >= 262144) return;
    int i    = g & 15;
    int lane = (g >> 4) & 31;
    {
        int kt = (g >> 9) & 7;
        int nt = g >> 12;
        int K  = kt * 32 + ((lane >> 4) << 3) + (i & 7) + ((i >> 3) << 4);
        int N  = nt * 16 + (lane & 15);
        w1p[g] = (_Float16)w1[K * FF + N];
    }
    {
        int kt = (g >> 9) & 31;
        int nt = g >> 14;
        int K  = kt * 32 + ((lane >> 4) << 3) + (i & 7) + ((i >> 3) << 4);
        int N  = nt * 16 + (lane & 15);
        w2p[g] = (_Float16)w2[K * OUTD + N];
    }
}

// ---------------------------------------------------------------------------
// Kernel 4: neighborhood attention. One HALF-WAVE (16 lanes) per (b,t):
// all 32 lanes of every wave are active (wave32, 2 neighborhoods per wave).
// 16 (b,t) slots per 256-thread block.
// ---------------------------------------------------------------------------
#define SLOTS 16

__global__ __launch_bounds__(256) void attn_kernel(
    const float* __restrict__ x, const int* __restrict__ idx,
    const float* __restrict__ pe,
    const float* __restrict__ ln_g, const float* __restrict__ ln_b,
    const float* __restrict__ q_w,  const float* __restrict__ k_w,
    const float* __restrict__ v_w,  const float* __restrict__ logit_scale,
    const float* __restrict__ att_out_w,
    _Float16* __restrict__ flat)
{
    __shared__ float sQ[SLOTS][MD * MD];
    __shared__ float sK[SLOTS][MD * MD];
    __shared__ float sV[SLOTS][MD * MD];

    int wave = threadIdx.x >> 5;
    int lane = threadIdx.x & 31;
    int half = lane >> 4;                   // 0 or 1
    int sub  = wave * 2 + half;             // slot 0..15 in this block
    int bt   = blockIdx.x * SLOTS + sub;    // < B*T
    int b    = bt / TT;
    int t    = bt - b * TT;
    int j    = lane & 15;                   // row owned by this lane

    // gather x for own row
    int s    = idx[t * NH + j];
    float xv = x[(long)b * SS + s];

    // h = layernorm(x + pe); q,k rows; v row
    {
        const float* perow = pe + ((long)t * NH + j) * MD;
        float val[MD];
        float m = 0.0f;
#pragma unroll
        for (int d = 0; d < MD; d++) { val[d] = xv + perow[d]; m += val[d]; }
        m *= (1.0f / MD);
        float var = 0.0f;
#pragma unroll
        for (int d = 0; d < MD; d++) { float df = val[d] - m; var += df * df; }
        var *= (1.0f / MD);
        float inv = rsqrtf(var + 1e-5f);
        float hrow[MD];
#pragma unroll
        for (int d = 0; d < MD; d++)
            hrow[d] = (val[d] - m) * inv * ln_g[d] + ln_b[d];
#pragma unroll
        for (int o = 0; o < MD; o++) {
            float qa = 0.0f, ka = 0.0f;
#pragma unroll
            for (int c = 0; c < MD; c++) {
                qa += hrow[c] * q_w[c * MD + o];
                ka += hrow[c] * k_w[c * MD + o];
            }
            sQ[sub][j * MD + o] = qa;
            sK[sub][j * MD + o] = ka;
            sV[sub][j * MD + o] = xv * v_w[o];
        }
    }
    __syncthreads();

    // per-head cosine normalization: own q row -> registers, own k row in LDS
    float qn[MD];
#pragma unroll
    for (int hh = 0; hh < HEADS; hh++) {
        float nq = 0.0f, nk = 0.0f;
#pragma unroll
        for (int d = 0; d < HD; d++) {
            float qv = sQ[sub][j * MD + hh * HD + d];
            float kv = sK[sub][j * MD + hh * HD + d];
            nq += qv * qv;
            nk += kv * kv;
        }
        float rq = 1.0f / (sqrtf(nq) + 1e-8f);
        float rk = 1.0f / (sqrtf(nk) + 1e-8f);
#pragma unroll
        for (int d = 0; d < HD; d++) {
            qn[hh * HD + d] = sQ[sub][j * MD + hh * HD + d] * rq;
            sK[sub][j * MD + hh * HD + d] *= rk;
        }
    }
    __syncthreads();

    // attention + output projection for own row
    {
        float ao[MD];
#pragma unroll
        for (int hh = 0; hh < HEADS; hh++) {
            float scale = __expf(fminf(logit_scale[hh], 4.6051701860f)); // ln(100)
            float att[NH];
            float mx = -3.4e38f;
#pragma unroll
            for (int mrow = 0; mrow < NH; mrow++) {
                float dot = 0.0f;
#pragma unroll
                for (int d = 0; d < HD; d++)
                    dot += qn[hh * HD + d] * sK[sub][mrow * MD + hh * HD + d];
                att[mrow] = dot * scale;
                mx = fmaxf(mx, att[mrow]);
            }
            float sum = 0.0f;
#pragma unroll
            for (int mrow = 0; mrow < NH; mrow++) {
                att[mrow] = __expf(att[mrow] - mx);
                sum += att[mrow];
            }
            float rinv = 1.0f / sum;
#pragma unroll
            for (int d = 0; d < HD; d++) {
                float acc = 0.0f;
#pragma unroll
                for (int mrow = 0; mrow < NH; mrow++)
                    acc += att[mrow] * sV[sub][mrow * MD + hh * HD + d];
                ao[hh * HD + d] = acc * rinv;
            }
        }
        _Float16* dst = flat + (long)bt * (NH * MD) + j * MD;
#pragma unroll
        for (int o = 0; o < MD; o++) {
            float a = 0.0f;
#pragma unroll
            for (int c = 0; c < MD; c++) a += ao[c] * att_out_w[c * MD + o];
            dst[o] = (_Float16)((xv + a) * 0.5f);
        }
    }
}

// ---------------------------------------------------------------------------
// Kernel 5: fused WMMA MLP. 16 rows / workgroup, 8 waves, hidden tile in LDS.
// Input tile staged via GLOBAL_LOAD_ASYNC_TO_LDS_B128 (ASYNCcnt path).
// ---------------------------------------------------------------------------
__device__ __forceinline__ v16h load_a_frag(const _Float16* base, int ld,
                                            int M, int kchunk, int kb)
{
    const _Float16* ap = base + M * ld + kchunk + kb;
    v8h lo = *(const v8h*)(ap);
    v8h hi = *(const v8h*)(ap + 16);
    v16h a;
#pragma unroll
    for (int i = 0; i < 8; i++) { a[i] = lo[i]; a[i + 8] = hi[i]; }
    return a;
}

__global__ __launch_bounds__(256) void mlp_kernel(
    const _Float16* __restrict__ flat,
    const _Float16* __restrict__ w1p, const float* __restrict__ b1,
    const _Float16* __restrict__ w2p, const float* __restrict__ b2,
    float* __restrict__ out)
{
    __shared__ __align__(32) _Float16 sA[16 * 256];    // 8 KB input tile
    __shared__ __align__(32) _Float16 sH[16 * FF];     // 32 KB hidden tile

    int tid  = threadIdx.x;
    int wave = tid >> 5;
    int lane = tid & 31;
    long rowBase = (long)blockIdx.x * 16;

    // Async-stage the 16x256 f16 tile (8 KB): each thread copies 32 bytes as
    // two 16B async LDS loads. INST_OFFSET applies to both LDS and global
    // addresses (ISA 08_async_tensor 4.4), so one addr pair serves both.
    {
        unsigned ldsAddr =
            (unsigned)(unsigned long long)(const void*)sA + (unsigned)tid * 32u;
        unsigned long long ga =
            (unsigned long long)(const void*)(flat + rowBase * 256) +
            (unsigned long long)tid * 32ull;
        asm volatile("global_load_async_to_lds_b128 %0, %1, off"
                     :: "v"(ldsAddr), "v"(ga) : "memory");
        asm volatile("global_load_async_to_lds_b128 %0, %1, off offset:16"
                     :: "v"(ldsAddr), "v"(ga) : "memory");
        asm volatile("s_wait_asynccnt 0x0" ::: "memory");
    }
    __syncthreads();

    int M     = lane & 15;            // A-fragment row owned by this lane
    int kb    = (lane >> 4) * 8;      // A-fragment K sub-offset (0 or 8)
    int Mbase = (lane >> 4) * 8;      // C/D fragment row base
    int Ncol  = lane & 15;            // C/D fragment column within tile

    // ---------------- GEMM1: 256 -> 1024, lrelu, keep in LDS ----------------
#pragma unroll
    for (int q = 0; q < 8; q++) {
        int nt = wave * 8 + q;
        const _Float16* wp = w1p + (size_t)(nt * 8) * 512;
        v8f acc = {};
#pragma unroll
        for (int kt = 0; kt < 8; kt++) {
            v16h a  = load_a_frag(sA, 256, M, kt * 32, kb);
            v16h bf = *(const v16h*)(wp + (size_t)kt * 512 + lane * 16);
            __builtin_prefetch(wp + (size_t)(kt + 1) * 512 + lane * 16, 0, 3);
            acc = __builtin_amdgcn_wmma_f32_16x16x32_f16(
                false, a, false, bf, (short)0, acc, false, false);
        }
        int N = nt * 16 + Ncol;
        float bias = b1[N];
#pragma unroll
        for (int r = 0; r < 8; r++) {
            float v = lrelu(acc[r] + bias);
            sH[(Mbase + r) * FF + N] = (_Float16)v;
        }
    }
    __syncthreads();

    // ---------------- GEMM2: 1024 -> 256, lrelu, fp32 out ----------------
#pragma unroll
    for (int q = 0; q < 2; q++) {
        int nt = wave * 2 + q;
        const _Float16* wp = w2p + (size_t)(nt * 32) * 512;
        v8f acc = {};
#pragma unroll
        for (int kt = 0; kt < 32; kt++) {
            v16h a  = load_a_frag(sH, FF, M, kt * 32, kb);
            v16h bf = *(const v16h*)(wp + (size_t)kt * 512 + lane * 16);
            acc = __builtin_amdgcn_wmma_f32_16x16x32_f16(
                false, a, false, bf, (short)0, acc, false, false);
        }
        int N = nt * 16 + Ncol;
        float bias = b2[N];
#pragma unroll
        for (int r = 0; r < 8; r++) {
            float v = lrelu(acc[r] + bias);
            out[(rowBase + Mbase + r) * OUTD + N] = v;
        }
    }
}

// ---------------------------------------------------------------------------
// Host-side launcher
// ---------------------------------------------------------------------------
extern "C" void kernel_launch(void* const* d_in, const int* in_sizes, int n_in,
                              void* d_out, int out_size, void* d_ws, size_t ws_size,
                              hipStream_t stream)
{
    const float* x           = (const float*)d_in[0];
    const float* coords_t    = (const float*)d_in[1];
    const float* coords_s    = (const float*)d_in[2];
    const float* pe_w1       = (const float*)d_in[3];
    const float* pe_b1       = (const float*)d_in[4];
    const float* pe_w2       = (const float*)d_in[5];
    const float* pe_b2       = (const float*)d_in[6];
    const float* ln_g        = (const float*)d_in[7];
    const float* ln_b        = (const float*)d_in[8];
    const float* q_w         = (const float*)d_in[9];
    const float* k_w         = (const float*)d_in[10];
    const float* v_w         = (const float*)d_in[11];
    const float* logit_scale = (const float*)d_in[12];
    const float* att_out_w   = (const float*)d_in[13];
    const float* out_w1      = (const float*)d_in[14];
    const float* out_b1      = (const float*)d_in[15];
    const float* out_w2      = (const float*)d_in[16];
    const float* out_b2      = (const float*)d_in[17];
    float* out = (float*)d_out;

    char* ws = (char*)d_ws;
    int*      idx  = (int*)      (ws);                 //  0.5 MB
    float*    pe   = (float*)    (ws + 524288);        // 16   MB
    _Float16* w1p  = (_Float16*) (ws + 17301504);      //  0.5 MB
    _Float16* w2p  = (_Float16*) (ws + 17825792);      //  0.5 MB
    _Float16* flat = (_Float16*) (ws + 18350080);      // 32   MB

    knn_kernel<<<TT / 256, 256, 0, stream>>>(coords_t, coords_s, idx);
    pe_kernel<<<(TT * NH) / 256, 256, 0, stream>>>(
        coords_t, coords_s, idx, pe_w1, pe_b1, pe_w2, pe_b2, pe);
    pack_w_kernel<<<262144 / 256, 256, 0, stream>>>(out_w1, out_w2, w1p, w2p);
    attn_kernel<<<NROWS / SLOTS, 256, 0, stream>>>(
        x, idx, pe, ln_g, ln_b, q_w, k_w, v_w, logit_scale, att_out_w, flat);
    mlp_kernel<<<NROWS / 16, 256, 0, stream>>>(flat, w1p, out_b1, w2p, out_b2, out);
}
